// PNAugment_4140348473509
// MI455X (gfx1250) — matrix operand order repaired and checked by reading, not compiled, not run on hardware
//
#include <hip/hip_runtime.h>

typedef __attribute__((ext_vector_type(16))) _Float16 v16h;
typedef __attribute__((ext_vector_type(8)))  _Float16 v8h;
typedef __attribute__((ext_vector_type(8)))  float    v8f;

#define N_ROWS 8192   // feat rows (output rows)
#define M_ROWS 4096   // P/N rows  (output cols)
#define D_DIM  64
#define LOG2E  1.44269504088896340736f

#define WMMA_F16(A, B, C) \
  __builtin_amdgcn_wmma_f32_16x16x32_f16(false, (A), false, (B), (short)0, (C), false, false)

// =====================================================================
// Prolog: fused fp32 -> (f16 hi, f16 lo residual) split + row sq-norms.
// Amortizes the conversion VALU over the 256-512x reuse of each row.
// =====================================================================
__global__ __launch_bounds__(256) void split_norm_kernel(const float* __restrict__ x,
                                                         _Float16* __restrict__ hiA,
                                                         _Float16* __restrict__ loA,
                                                         float* __restrict__ norms,
                                                         int rows) {
  int r = blockIdx.x * blockDim.x + threadIdx.x;
  if (r >= rows) return;
  const float4* src = reinterpret_cast<const float4*>(x) + (size_t)r * (D_DIM / 4);
  v8h* hd = reinterpret_cast<v8h*>(hiA + (size_t)r * D_DIM);
  v8h* ld = reinterpret_cast<v8h*>(loA + (size_t)r * D_DIM);
  float s = 0.f;
#pragma unroll
  for (int c = 0; c < D_DIM / 8; ++c) {
    float4 t0 = src[2 * c], t1 = src[2 * c + 1];
    float b[8] = {t0.x, t0.y, t0.z, t0.w, t1.x, t1.y, t1.z, t1.w};
    v8h h, l;
#pragma unroll
    for (int j = 0; j < 8; ++j) {
      float v = b[j];
      s += v * v;
      _Float16 hh = (_Float16)v;
      h[j] = hh;
      l[j] = (_Float16)(v - (float)hh);   // exact residual, then rounded: x ~= h+l, err ~2^-22
    }
    hd[c] = h;
    ld[c] = l;
  }
  norms[r] = s;
}

__device__ __forceinline__ v16h cat8(v8h a, v8h b) {
  return __builtin_shufflevector(a, b, 0, 1, 2, 3, 4, 5, 6, 7, 8, 9, 10, 11, 12, 13, 14, 15);
}

// =====================================================================
// Fast path: inputs pre-split to f16 hi/lo. Block = 256 thr = 8 waves
// arranged 2(M) x 4(N); each wave owns a 64x16 strip (4 m-tiles),
// B fragments register-resident.
// =====================================================================
__global__ __launch_bounds__(256) void pn_prob_f16_kernel(
    const _Float16* __restrict__ fH, const _Float16* __restrict__ fL,
    const _Float16* __restrict__ pH, const _Float16* __restrict__ pL,
    const _Float16* __restrict__ nH, const _Float16* __restrict__ nL,
    const float* __restrict__ F2, const float* __restrict__ P2,
    const float* __restrict__ N2, float* __restrict__ out) {
  const int lane = threadIdx.x & 31;
  const int wave = threadIdx.x >> 5;   // 0..7
  const int wm   = wave >> 2;          // 0..1
  const int wn   = wave & 3;           // 0..3
  const int g    = lane >> 4;          // lane group
  const int lr   = lane & 15;

  const int nBase = blockIdx.x * 64 + wn * 16;
  const int mBase = blockIdx.y * 128 + wm * 64;

  // B fragments: lane = column N=lr, K run [16g+32kk, +16) -> two b128 loads each
  v16h bPhi[2], bPlo[2], bNhi[2], bNlo[2];
  {
    const size_t boff = (size_t)(nBase + lr) * D_DIM;
#pragma unroll
    for (int kk = 0; kk < 2; ++kk) {
      const size_t o = boff + kk * 32 + g * 16;
      bPhi[kk] = cat8(*(const v8h*)(pH + o), *(const v8h*)(pH + o + 8));
      bPlo[kk] = cat8(*(const v8h*)(pL + o), *(const v8h*)(pL + o + 8));
      bNhi[kk] = cat8(*(const v8h*)(nH + o), *(const v8h*)(nH + o + 8));
      bNlo[kk] = cat8(*(const v8h*)(nL + o), *(const v8h*)(nL + o + 8));
    }
  }
  const float p2v = P2[nBase + lr];
  const float n2v = N2[nBase + lr];

#pragma unroll
  for (int mt = 0; mt < 4; ++mt) {
    const int mB = mBase + mt * 16;
    const size_t aoff = (size_t)(mB + lr) * D_DIM;
    if (mt < 3) __builtin_prefetch(fH + aoff + 16 * D_DIM);

    // A fragments: lane = row M=lr, K run [16g,+16) per K-half
    v16h aHi[2], aLo[2];
#pragma unroll
    for (int kk = 0; kk < 2; ++kk) {
      const size_t o = aoff + kk * 32 + 16 * g;
      aHi[kk] = cat8(*(const v8h*)(fH + o), *(const v8h*)(fH + o + 8));
      aLo[kk] = cat8(*(const v8h*)(fL + o), *(const v8h*)(fL + o + 8));
    }

    // F2 for rows mB+8g .. +8 : two b128 loads
    const float4* f2p = reinterpret_cast<const float4*>(F2 + mB + 8 * g);
    float4 f2a = f2p[0], f2b = f2p[1];
    const float f2v[8] = {f2a.x, f2a.y, f2a.z, f2a.w, f2b.x, f2b.y, f2b.z, f2b.w};

    v8f accP = {}, accN = {};
#pragma unroll
    for (int kk = 0; kk < 2; ++kk) {
      accP = WMMA_F16(aHi[kk], bPhi[kk], accP);
      accP = WMMA_F16(aHi[kk], bPlo[kk], accP);
      accP = WMMA_F16(aLo[kk], bPhi[kk], accP);
      accN = WMMA_F16(aHi[kk], bNhi[kk], accN);
      accN = WMMA_F16(aHi[kk], bNlo[kk], accN);
      accN = WMMA_F16(aLo[kk], bNhi[kk], accN);
    }

    float* orow = out + (size_t)(mB + 8 * g) * M_ROWS + nBase + lr;
#pragma unroll
    for (int r = 0; r < 8; ++r) {
      const float f2  = f2v[r];
      const float sqp = fmaxf(__builtin_fmaf(accP[r], -2.f, f2 + p2v), 0.f);
      const float sqn = fmaxf(__builtin_fmaf(accN[r], -2.f, f2 + n2v), 0.f);
      const float z   = __builtin_amdgcn_sqrtf(sqp) - __builtin_amdgcn_sqrtf(sqn); // pos-neg
      const float e   = __builtin_amdgcn_exp2f(z * LOG2E);
      orow[(size_t)r * M_ROWS] = __builtin_amdgcn_rcpf(1.f + e);
    }
  }
}

// =====================================================================
// Fallback path (small ws): split in-kernel (round-1 style), norms only in ws
// =====================================================================
__global__ __launch_bounds__(256) void row_sqnorm_kernel(const float* __restrict__ x,
                                                         float* __restrict__ out, int rows) {
  int r = blockIdx.x * blockDim.x + threadIdx.x;
  if (r >= rows) return;
  const float4* p = reinterpret_cast<const float4*>(x) + (size_t)r * (D_DIM / 4);
  float s = 0.f;
#pragma unroll
  for (int i = 0; i < D_DIM / 4; ++i) {
    float4 v = p[i];
    s += v.x * v.x + v.y * v.y + v.z * v.z + v.w * v.w;
  }
  out[r] = s;
}

__device__ __forceinline__ void load_split16(const float* __restrict__ src, v16h& hi, v16h& lo) {
  const float4* s4 = reinterpret_cast<const float4*>(src);
  float buf[16];
#pragma unroll
  for (int i = 0; i < 4; ++i) {
    float4 t = s4[i];
    buf[4 * i + 0] = t.x; buf[4 * i + 1] = t.y; buf[4 * i + 2] = t.z; buf[4 * i + 3] = t.w;
  }
#pragma unroll
  for (int j = 0; j < 16; ++j) {
    float x = buf[j];
    _Float16 h = (_Float16)x;
    hi[j] = h;
    lo[j] = (_Float16)(x - (float)h);
  }
}

__device__ __forceinline__ void load_split8(const float* __restrict__ src, v16h& hi, v16h& lo, int off) {
  const float4* s4 = reinterpret_cast<const float4*>(src);
  float buf[8];
#pragma unroll
  for (int i = 0; i < 2; ++i) {
    float4 t = s4[i];
    buf[4 * i + 0] = t.x; buf[4 * i + 1] = t.y; buf[4 * i + 2] = t.z; buf[4 * i + 3] = t.w;
  }
#pragma unroll
  for (int j = 0; j < 8; ++j) {
    float x = buf[j];
    _Float16 h = (_Float16)x;
    hi[off + j] = h;
    lo[off + j] = (_Float16)(x - (float)h);
  }
}

__global__ __launch_bounds__(256) void pn_prob_fallback_kernel(
    const float* __restrict__ feat, const float* __restrict__ Pm, const float* __restrict__ Nm,
    const float* __restrict__ F2, const float* __restrict__ P2, const float* __restrict__ N2,
    float* __restrict__ out) {
  const int lane = threadIdx.x & 31;
  const int wave = threadIdx.x >> 5;
  const int wm = wave >> 2, wn = wave & 3;
  const int g = lane >> 4, lr = lane & 15;
  const int nBase = blockIdx.x * 64 + wn * 16;
  const int mBase = blockIdx.y * 128 + wm * 64;

  v16h bPhi[2], bPlo[2], bNhi[2], bNlo[2];
  {
    const float* prow = Pm + (size_t)(nBase + lr) * D_DIM;
    const float* nrow = Nm + (size_t)(nBase + lr) * D_DIM;
#pragma unroll
    for (int kk = 0; kk < 2; ++kk) {
      const int base = kk * 32 + g * 16;
      load_split16(prow + base, bPhi[kk], bPlo[kk]);
      load_split16(nrow + base, bNhi[kk], bNlo[kk]);
    }
  }
  const float p2v = P2[nBase + lr];
  const float n2v = N2[nBase + lr];

#pragma unroll
  for (int mt = 0; mt < 4; ++mt) {
    const int mB = mBase + mt * 16;
    const float* arow = feat + (size_t)(mB + lr) * D_DIM;
    v16h aHi[2], aLo[2];
#pragma unroll
    for (int kk = 0; kk < 2; ++kk) {
      const int b0 = kk * 32 + 16 * g;
      load_split8(arow + b0, aHi[kk], aLo[kk], 0);
      load_split8(arow + b0 + 8, aHi[kk], aLo[kk], 8);
    }
    v8f accP = {}, accN = {};
#pragma unroll
    for (int kk = 0; kk < 2; ++kk) {
      accP = WMMA_F16(aHi[kk], bPhi[kk], accP);
      accP = WMMA_F16(aHi[kk], bPlo[kk], accP);
      accP = WMMA_F16(aLo[kk], bPhi[kk], accP);
      accN = WMMA_F16(aHi[kk], bNhi[kk], accN);
      accN = WMMA_F16(aHi[kk], bNlo[kk], accN);
      accN = WMMA_F16(aLo[kk], bNhi[kk], accN);
    }
#pragma unroll
    for (int r = 0; r < 8; ++r) {
      const int row = mB + r + 8 * g;
      const float f2  = F2[row];
      const float sqp = fmaxf(__builtin_fmaf(accP[r], -2.f, f2 + p2v), 0.f);
      const float sqn = fmaxf(__builtin_fmaf(accN[r], -2.f, f2 + n2v), 0.f);
      const float z   = __builtin_amdgcn_sqrtf(sqp) - __builtin_amdgcn_sqrtf(sqn);
      out[(size_t)row * M_ROWS + nBase + lr] = __builtin_amdgcn_rcpf(1.f + __builtin_amdgcn_exp2f(z * LOG2E));
    }
  }
}

extern "C" void kernel_launch(void* const* d_in, const int* in_sizes, int n_in,
                              void* d_out, int out_size, void* d_ws, size_t ws_size,
                              hipStream_t stream) {
  (void)in_sizes; (void)n_in; (void)out_size;
  const float* feat = (const float*)d_in[0];
  const float* Pm   = (const float*)d_in[1];
  const float* Nm   = (const float*)d_in[2];
  float* out = (float*)d_out;

  // ws layout: [F2 8192 f32][P2 4096][N2 4096] then f16 hi/lo arrays
  float* F2 = (float*)d_ws;
  float* P2 = F2 + N_ROWS;
  float* N2 = P2 + M_ROWS;
  _Float16* fH = (_Float16*)((char*)d_ws + 65536);
  _Float16* fL = fH + (size_t)N_ROWS * D_DIM;
  _Float16* pH = fL + (size_t)N_ROWS * D_DIM;
  _Float16* pL = pH + (size_t)M_ROWS * D_DIM;
  _Float16* nH = pL + (size_t)M_ROWS * D_DIM;
  _Float16* nL = nH + (size_t)M_ROWS * D_DIM;
  const size_t need = 65536 + (size_t)(2 * N_ROWS + 4 * M_ROWS) * D_DIM * sizeof(_Float16);

  if (ws_size >= need) {
    split_norm_kernel<<<dim3(N_ROWS / 256), dim3(256), 0, stream>>>(feat, fH, fL, F2, N_ROWS);
    split_norm_kernel<<<dim3(M_ROWS / 256), dim3(256), 0, stream>>>(Pm, pH, pL, P2, M_ROWS);
    split_norm_kernel<<<dim3(M_ROWS / 256), dim3(256), 0, stream>>>(Nm, nH, nL, N2, M_ROWS);
    pn_prob_f16_kernel<<<dim3(M_ROWS / 64, N_ROWS / 128), dim3(256), 0, stream>>>(
        fH, fL, pH, pL, nH, nL, F2, P2, N2, out);
  } else {
    row_sqnorm_kernel<<<dim3(N_ROWS / 256), dim3(256), 0, stream>>>(feat, F2, N_ROWS);
    row_sqnorm_kernel<<<dim3(M_ROWS / 256), dim3(256), 0, stream>>>(Pm, P2, M_ROWS);
    row_sqnorm_kernel<<<dim3(M_ROWS / 256), dim3(256), 0, stream>>>(Nm, N2, M_ROWS);
    pn_prob_fallback_kernel<<<dim3(M_ROWS / 64, N_ROWS / 128), dim3(256), 0, stream>>>(
        feat, Pm, Nm, F2, P2, N2, out);
  }
}
